// ODE2cVAERegression_27771258536544
// MI455X (gfx1250) — matrix operand order repaired and checked by reading, not compiled, not run on hardware
//
#include <hip/hip_runtime.h>

#define N_    2048
#define T_    50
#define C_    10
#define QS_   16
#define QO_   8
#define H_    50
#define EH_   128
#define HD1_  256
#define HD2_  128
#define DT_   0.1f
#define LOG2PI_ 1.8378770664093453f
#define NDATA_ (10 * N_)

#define ODE_WAVES 8          // one wave per JVP direction
#define USE_ASYNC_LDS 1      // global_load_async_to_lds_b128 path in decoder

typedef __attribute__((ext_vector_type(16))) _Float16 v16h;
typedef __attribute__((ext_vector_type(8)))  float    v8f;
typedef __attribute__((ext_vector_type(8)))  _Float16 h8;

#define WMMA_F16(a, b, c) \
  __builtin_amdgcn_wmma_f32_16x16x32_f16(false, (a), false, (b), (short)0, (c), false, false)

// compiler fence + wave-level scheduling barrier: orders same-wave cross-lane
// LDS produce->consume (DS ops are issued in-order within a wave on CDNA5)
__device__ __forceinline__ void wfence() {
  __asm__ volatile("" ::: "memory");
  __builtin_amdgcn_wave_barrier();
}

#if USE_ASYNC_LDS
__device__ __forceinline__ void async_ld_b128(void* ldsDst, const void* gsrc) {
  unsigned lo = (unsigned)(size_t)ldsDst;   // low 32 bits of generic LDS ptr = LDS byte addr
  asm volatile("global_load_async_to_lds_b128 %0, %1, off"
               :: "v"(lo), "v"(gsrc) : "memory");
}
__device__ __forceinline__ void wait_async0() {
  asm volatile("s_wait_asynccnt 0x0" ::: "memory");
}
#endif

// ---------------- fragment packing from LDS (f16, padded, 16B-aligned runs) ---
__device__ __forceinline__ v16h ldsFragA(const _Float16* base, int ldk, int k0, int lane) {
  int row = lane & 15;
  int k   = k0 + ((lane & 16) ? 8 : 0);
  const _Float16* p = base + row * ldk + k;
  h8 lo = *(const h8*)(p);
  h8 hi = *(const h8*)(p + 16);
  v16h r;
#pragma unroll
  for (int i = 0; i < 8; ++i) { r[i] = lo[i]; r[i + 8] = hi[i]; }
  return r;
}

__device__ __forceinline__ v16h ldsFragB(const _Float16* baseT, int ldk, int k0, int lane) {
  int n  = lane & 15;
  int kb = k0 + ((lane & 16) ? 8 : 0);
  const _Float16* p = baseT + n * ldk + kb;
  h8 lo = *(const h8*)(p);
  h8 hi = *(const h8*)(p + 16);
  v16h r;
#pragma unroll
  for (int i = 0; i < 8; ++i) { r[i] = lo[i]; r[i + 8] = hi[i]; }
  return r;
}

__device__ __forceinline__ void mm16_f32(const _Float16* A, int lda,
                                         const _Float16* Bt, int ldb,
                                         float* Co, int ldc,
                                         int K, int Nout, int lane, const float* bias) {
  for (int n0 = 0; n0 < Nout; n0 += 16) {
    v8f acc = {};
    for (int k0 = 0; k0 < K; k0 += 32) {
      v16h a = ldsFragA(A, lda, k0, lane);
      v16h b = ldsFragB(Bt + n0 * ldb, ldb, k0, lane);
      acc = WMMA_F16(a, b, acc);
    }
    int n  = n0 + (lane & 15);
    int mB = (lane & 16) ? 8 : 0;
    float bv = bias ? bias[n] : 0.f;
#pragma unroll
    for (int r = 0; r < 8; ++r) Co[(mB + r) * ldc + n] = acc[r] + bv;
  }
}

__device__ __forceinline__ void mm16_relu_h(const _Float16* A, int lda,
                                            const _Float16* Bt, int ldb,
                                            _Float16* Co, int ldc,
                                            int K, int Nout, int lane, const float* bias) {
  for (int n0 = 0; n0 < Nout; n0 += 16) {
    v8f acc = {};
    for (int k0 = 0; k0 < K; k0 += 32) {
      v16h a = ldsFragA(A, lda, k0, lane);
      v16h b = ldsFragB(Bt + n0 * ldb, ldb, k0, lane);
      acc = WMMA_F16(a, b, acc);
    }
    int n  = n0 + (lane & 15);
    int mB = (lane & 16) ? 8 : 0;
    float bv = bias ? bias[n] : 0.f;
#pragma unroll
    for (int r = 0; r < 8; ++r) {
      float v = acc[r] + bv;
      Co[(mB + r) * ldc + n] = (_Float16)(v > 0.f ? v : 0.f);
    }
  }
}

// W[K][Nout] f32 -> Wt[NPad][KPad] f16 transposed, zero padded
__device__ __forceinline__ void loadWt(const float* W, int K, int Nout,
                                       _Float16* Wt, int KPad, int NPad,
                                       int tid, int nthr) {
  for (int i = tid; i < NPad * KPad; i += nthr) {
    int n = i / KPad, k = i - n * KPad;
    float v = (n < Nout && k < K) ? W[k * Nout + n] : 0.f;
    Wt[i] = (_Float16)v;
  }
}
__device__ __forceinline__ void loadBias(const float* b, int Nv, float* out, int NPad,
                                         int tid, int nthr) {
  for (int i = tid; i < NPad; i += nthr) out[i] = (i < Nv) ? b[i] : 0.f;
}

// ---------------- celu + LayerNorm forward (saves x̂, celu', 1/σ) -------------
__device__ __forceinline__ void celu_ln_fwd(const float* pre, _Float16* xhat,
                                            _Float16* dact, float* istd, int lane) {
  if (lane < 16) {
    const float* p = pre + lane * 64;
    _Float16* dx = dact + lane * 64;
    _Float16* xo = xhat + lane * 64;
    float a[H_]; float s = 0.f;
    for (int j = 0; j < H_; ++j) {
      float x = p[j];
      float c = x > 0.f ? x : expm1f(x);
      float d = x > 0.f ? 1.f : __expf(x);
      a[j] = c; s += c;
      dx[j] = (_Float16)d;
    }
    float m = s * (1.f / H_);
    float v = 0.f;
    for (int j = 0; j < H_; ++j) { float t = a[j] - m; v += t * t; }
    v *= (1.f / H_);
    float rs = __frsqrt_rn(v + 1e-5f);
    istd[lane] = rs;
    for (int j = 0; j < H_; ++j) xo[j] = (_Float16)((a[j] - m) * rs);
    for (int j = H_; j < 64; ++j) { xo[j] = (_Float16)0.f; dx[j] = (_Float16)0.f; }
  }
}

// tangent through celu' then LN-JVP
__device__ __forceinline__ void ln_jvp(const float* tpre, const _Float16* dact,
                                       const _Float16* xhat, const float* istd,
                                       _Float16* tnext, int lane) {
  if (lane < 16) {
    const float* tp = tpre + lane * 64;
    const _Float16* dx = dact + lane * 64;
    const _Float16* xh = xhat + lane * 64;
    _Float16* to = tnext + lane * 64;
    float u[H_]; float su = 0.f, sxu = 0.f;
    for (int j = 0; j < H_; ++j) {
      float uu = tp[j] * (float)dx[j];
      u[j] = uu; su += uu; sxu += uu * (float)xh[j];
    }
    float mu = su * (1.f / H_), mxu = sxu * (1.f / H_), rs = istd[lane];
    for (int j = 0; j < H_; ++j) to[j] = (_Float16)((u[j] - mu - (float)xh[j] * mxu) * rs);
    for (int j = H_; j < 64; ++j) to[j] = (_Float16)0.f;
  }
}

// =================== prep ======================================================
__global__ void zero_acc_kernel(float* accum) {
  if (threadIdx.x < 2) accum[threadIdx.x] = 0.f;
}

__global__ void prep_wd2_kernel(const float* Wd2, _Float16* wd2t) {
  int i = blockIdx.x * 256 + threadIdx.x;
  if (i < HD2_ * HD1_) {
    int n = i / HD1_, k = i - n * HD1_;
    wd2t[i] = (_Float16)Wd2[(size_t)k * HD2_ + n];
  }
}

// =================== encoder (16 samples / wave) ===============================
__global__ __launch_bounds__(32)
void enc_kernel(const float* X, const float* cats, const float* eps,
                const float* We1, const float* be1, const float* Wm, const float* bm,
                const float* Wv, const float* bv, const float* Wode, const float* bode,
                float* zt, float* logp) {
  __shared__ _Float16 We1t[EH_][32];
  __shared__ _Float16 Wmt[16][128];
  __shared__ _Float16 Wvt[16][128];
  __shared__ _Float16 Wodet[16][32];
  __shared__ float sbe1[EH_], sbm[16], sbv[16], sbode[16];
  __shared__ _Float16 x0a[16][32];
  __shared__ _Float16 hA[16][EH_];
  __shared__ float qm[16][16], qv[16][16];
  __shared__ _Float16 z0a[16][32];
  __shared__ float zode[16][16];

  int lane = threadIdx.x;
  int tile = blockIdx.x;
  loadWt(We1, 1 + C_, EH_, &We1t[0][0], 32, EH_, lane, 32);
  loadWt(Wm, EH_, QS_, &Wmt[0][0], 128, 16, lane, 32);
  loadWt(Wv, EH_, QS_, &Wvt[0][0], 128, 16, lane, 32);
  loadWt(Wode, QS_ + C_, 2 * QO_, &Wodet[0][0], 32, 16, lane, 32);
  loadBias(be1, EH_, sbe1, EH_, lane, 32);
  loadBias(bm, QS_, sbm, 16, lane, 32);
  loadBias(bv, QS_, sbv, 16, lane, 32);
  loadBias(bode, 2 * QO_, sbode, 16, lane, 32);
  for (int i = lane; i < 16 * 32; i += 32) {
    int m = i >> 5, k = i & 31;
    int n = tile * 16 + m;
    float v = 0.f;
    if (k == 0)      v = X[(size_t)n * T_];
    else if (k < 11) v = cats[((size_t)n * T_) * C_ + (k - 1)];
    x0a[m][k] = (_Float16)v;
  }
  __syncthreads();
  mm16_relu_h(&x0a[0][0], 32, &We1t[0][0], 32, &hA[0][0], EH_, 32, EH_, lane, sbe1);
  __syncthreads();
  mm16_f32(&hA[0][0], EH_, &Wmt[0][0], 128, &qm[0][0], 16, 128, 16, lane, sbm);
  mm16_f32(&hA[0][0], EH_, &Wvt[0][0], 128, &qv[0][0], 16, 128, 16, lane, sbv);
  __syncthreads();
  if (lane < 16) {
    int n = tile * 16 + lane;
    float se = 0.f;
    for (int j = 0; j < QS_; ++j) {
      float e = eps[(size_t)n * QS_ + j]; se += e * e;
      float z = qm[lane][j] + __expf(0.5f * qv[lane][j]) * e;
      z0a[lane][j] = (_Float16)z;
    }
    for (int j = 0; j < C_; ++j) z0a[lane][QS_ + j] = (_Float16)cats[((size_t)n * T_) * C_ + j];
    for (int j = QS_ + C_; j < 32; ++j) z0a[lane][j] = (_Float16)0.f;
    logp[(size_t)n * T_] = -0.5f * se - 0.5f * QS_ * LOG2PI_;
  }
  __syncthreads();
  mm16_f32(&z0a[0][0], 32, &Wodet[0][0], 32, &zode[0][0], 16, 32, 16, lane, sbode);
  __syncthreads();
  if (lane < 16) {
    int n = tile * 16 + lane;
    for (int j = 0; j < 16; ++j) zt[((size_t)n * T_) * 16 + j] = zode[lane][j];
  }
}

// =================== RK4 neural-ODE: 8 waves, 1 JVP direction per wave =========
__global__ __launch_bounds__(32 * ODE_WAVES)
void ode_kernel(const float* w1, const float* b1, const float* w2, const float* b2,
                const float* w3, const float* b3, const float* w4, const float* b4,
                float* zt, float* logp) {
  extern __shared__ char smem_raw[];
  char* p = (char*)(((size_t)smem_raw + 63) & ~(size_t)63);
  auto alloc = [&](size_t bytes) -> char* {
    char* r = p; p += (bytes + 63) & ~(size_t)63; return r;
  };
  _Float16* w1t  = (_Float16*)alloc(64 * 32 * 2);
  _Float16* w2t  = (_Float16*)alloc(64 * 64 * 2);
  _Float16* w3t  = (_Float16*)alloc(64 * 64 * 2);
  _Float16* w4t  = (_Float16*)alloc(16 * 64 * 2);
  float* sb1 = (float*)alloc(64 * 4);
  float* sb2 = (float*)alloc(64 * 4);
  float* sb3 = (float*)alloc(64 * 4);
  float* sb4 = (float*)alloc(16 * 4);
  _Float16* vsA  = (_Float16*)alloc(16 * 32 * 2);
  float*    pre0 = (float*)alloc(16 * 64 * 4);
  _Float16* xh0  = (_Float16*)alloc(16 * 64 * 2);
  _Float16* xh1  = (_Float16*)alloc(16 * 64 * 2);
  _Float16* xh2  = (_Float16*)alloc(16 * 64 * 2);
  _Float16* da0  = (_Float16*)alloc(16 * 64 * 2);
  _Float16* da1  = (_Float16*)alloc(16 * 64 * 2);
  _Float16* da2  = (_Float16*)alloc(16 * 64 * 2);
  float* istd  = (float*)alloc(3 * 16 * 4);
  float* fout  = (float*)alloc(16 * 16 * 4);
  float* trbuf = (float*)alloc(ODE_WAVES * 16 * 4);
  _Float16* tactAll = (_Float16*)alloc(ODE_WAVES * 16 * 64 * 2);
  float*    preAll  = (float*)alloc(ODE_WAVES * 16 * 64 * 4);
  float*    toutAll = (float*)alloc(ODE_WAVES * 16 * 16 * 4);

  int tid = threadIdx.x, lane = tid & 31, wave = tid >> 5;
  int tile = blockIdx.x;
  loadWt(w1, 2 * QO_, H_, w1t, 32, 64, tid, blockDim.x);
  loadWt(w2, H_, H_, w2t, 64, 64, tid, blockDim.x);
  loadWt(w3, H_, H_, w3t, 64, 64, tid, blockDim.x);
  loadWt(w4, H_, QO_, w4t, 64, 16, tid, blockDim.x);
  loadBias(b1, H_, sb1, 64, tid, blockDim.x);
  loadBias(b2, H_, sb2, 64, tid, blockDim.x);
  loadBias(b3, H_, sb3, 64, tid, blockDim.x);
  loadBias(b4, QO_, sb4, 16, tid, blockDim.x);
  __syncthreads();

  _Float16* tact  = tactAll + wave * 16 * 64;
  float*    prew  = preAll  + wave * 16 * 64;
  float*    toutw = toutAll + wave * 16 * 16;

  float v0[16]; float lp = 0.f;
  if (wave == 0 && lane < 16) {
    int n = tile * 16 + lane;
    for (int j = 0; j < 16; ++j) v0[j] = zt[((size_t)n * T_) * 16 + j];
    lp = logp[(size_t)n * T_];
  }

  for (int t = 0; t < T_ - 1; ++t) {
    float acc[16]; float cacc = 0.f;
    float aprev[16]; float vin[16];
    if (wave == 0 && lane < 16) for (int j = 0; j < 16; ++j) acc[j] = 0.f;
    for (int s = 0; s < 4; ++s) {
      float cin = (s == 0) ? 0.f : ((s == 3) ? 1.f : 0.5f);
      float wgt = (s == 1 || s == 2) ? 2.f : 1.f;
      if (wave == 0 && lane < 16) {
        for (int j = 0; j < 16; ++j) {
          vin[j] = (s == 0) ? v0[j] : v0[j] + cin * DT_ * aprev[j];
          vsA[lane * 32 + j] = (_Float16)vin[j];
        }
        for (int j = 16; j < 32; ++j) vsA[lane * 32 + j] = (_Float16)0.f;
      }
      __syncthreads();
      // ---- base forward: wave 0 only (DS in-order within wave; wfence stops reordering)
      if (wave == 0) {
        mm16_f32(vsA, 32, w1t, 32, pre0, 64, 32, 64, lane, sb1);      wfence();
        celu_ln_fwd(pre0, xh0, da0, istd + 0, lane);                  wfence();
        mm16_f32(xh0, 64, w2t, 64, pre0, 64, 64, 64, lane, sb2);      wfence();
        celu_ln_fwd(pre0, xh1, da1, istd + 16, lane);                 wfence();
        mm16_f32(xh1, 64, w3t, 64, pre0, 64, 64, 64, lane, sb3);      wfence();
        celu_ln_fwd(pre0, xh2, da2, istd + 32, lane);                 wfence();
        mm16_f32(xh2, 64, w4t, 64, fout, 16, 64, 16, lane, sb4);
      }
      __syncthreads();
      // ---- tangent pass: direction d == wave
      {
        int d = wave;
        if (lane < 16) {
          const _Float16* dx = da0 + lane * 64;
          const _Float16* x1 = xh0 + lane * 64;
          float u[H_]; float su = 0.f, sxu = 0.f;
          for (int j = 0; j < H_; ++j) {
            float t0 = (float)w1t[j * 32 + d];          // = w1[d][j]
            float uu = t0 * (float)dx[j];
            u[j] = uu; su += uu; sxu += uu * (float)x1[j];
          }
          float mu = su * (1.f / H_), mxu = sxu * (1.f / H_), rs = istd[lane];
          for (int j = 0; j < H_; ++j)
            tact[lane * 64 + j] = (_Float16)((u[j] - mu - (float)x1[j] * mxu) * rs);
          for (int j = H_; j < 64; ++j) tact[lane * 64 + j] = (_Float16)0.f;
        }
        wfence();
        mm16_f32(tact, 64, w2t, 64, prew, 64, 64, 64, lane, nullptr); wfence();
        ln_jvp(prew, da1, xh1, istd + 16, tact, lane);                wfence();
        mm16_f32(tact, 64, w3t, 64, prew, 64, 64, 64, lane, nullptr); wfence();
        ln_jvp(prew, da2, xh2, istd + 32, tact, lane);                wfence();
        mm16_f32(tact, 64, w4t, 64, toutw, 16, 64, 16, lane, nullptr); wfence();
        if (lane < 16) trbuf[wave * 16 + lane] = toutw[lane * 16 + d];
      }
      __syncthreads();
      // ---- RK4 gather on wave 0
      if (wave == 0 && lane < 16) {
        float trm = 0.f;
        for (int w = 0; w < ODE_WAVES; ++w) trm += trbuf[w * 16 + lane];
        for (int j = 0; j < 8; ++j)  aprev[j] = fout[lane * 16 + j];
        for (int j = 8; j < 16; ++j) aprev[j] = vin[j - 8];
        for (int j = 0; j < 16; ++j) acc[j] += wgt * aprev[j];
        cacc += wgt * (-trm);
      }
      __syncthreads();
    }
    if (wave == 0 && lane < 16) {
      int n = tile * 16 + lane;
      for (int j = 0; j < 16; ++j) v0[j] += (DT_ / 6.f) * acc[j];
      lp += (DT_ / 6.f) * cacc;
      for (int j = 0; j < 16; ++j) zt[((size_t)n * T_ + t + 1) * 16 + j] = v0[j];
      logp[(size_t)n * T_ + t + 1] = lp;
    }
    __syncthreads();
  }
}

// =================== decoder: async double-buffered Wd2 K-slabs ================
__global__ __launch_bounds__(32)
void dec_kernel(const float* X, const float* cats, const float* mask,
                const float* Wd1, const float* bd1, const float* bd2,
                const float* Wd3, const float* bd3,
                const _Float16* wd2t_g, const float* zt,
                float* xrec, float* accum) {
  __shared__ _Float16 wd1t[HD1_][32];
  __shared__ _Float16 slab[2][HD2_][64];
  __shared__ float sbd1[HD1_], sbd2[HD2_], swd3[HD2_];
  __shared__ _Float16 ain[16][32];
  __shared__ _Float16 h1[16][HD1_];
  __shared__ _Float16 h2[16][HD2_];
  __shared__ float red[16];

  int lane = threadIdx.x;
  int tile = blockIdx.x;
  __builtin_prefetch(wd2t_g + lane * 64, 0, 1);        // global_prefetch_b8
  loadWt(Wd1, QO_ + C_, HD1_, &wd1t[0][0], 32, HD1_, lane, 32);
  loadBias(bd1, HD1_, sbd1, HD1_, lane, 32);
  loadBias(bd2, HD2_, sbd2, HD2_, lane, 32);
  for (int i = lane; i < HD2_; i += 32) swd3[i] = Wd3[i];
  for (int i = lane; i < 16 * 32; i += 32) {
    int m = i >> 5, k = i & 31;
    int r = tile * 16 + m;
    int n = r / T_, t = r - n * T_;
    float v = 0.f;
    if (k < QO_)           v = zt[((size_t)n * T_ + t) * 16 + QO_ + k];
    else if (k < QO_ + C_) v = cats[((size_t)n * T_) * C_ + (k - QO_)];
    ain[m][k] = (_Float16)v;
  }

  auto issueSlab = [&](int buf, int ks) {
#if USE_ASYNC_LDS
    for (int i = lane; i < HD2_ * 8; i += 32) {        // 1024 x 16B chunks
      int nrow = i >> 3, kin = (i & 7) * 8;
      async_ld_b128(&slab[buf][nrow][kin], wd2t_g + (size_t)nrow * HD1_ + ks + kin);
    }
#else
    const unsigned int* src = (const unsigned int*)wd2t_g;
    unsigned int* dst = (unsigned int*)&slab[buf][0][0];
    for (int i = lane; i < HD2_ * 32; i += 32) {
      int nrow = i >> 5, kin = i & 31;
      dst[nrow * 32 + kin] = src[nrow * 128 + (ks >> 1) + kin];
    }
#endif
  };
  issueSlab(0, 0);
  __syncthreads();
  mm16_relu_h(&ain[0][0], 32, &wd1t[0][0], 32, &h1[0][0], HD1_, 32, HD1_, lane, sbd1);
  __syncthreads();

  v8f acc[8] = {};
  for (int ks = 0; ks < HD1_; ks += 64) {
    int buf = (ks >> 6) & 1;
#if USE_ASYNC_LDS
    wait_async0();                                     // s_wait_asynccnt 0
#endif
    __syncthreads();
    if (ks + 64 < HD1_) issueSlab(buf ^ 1, ks + 64);   // prefetch next slab async
#pragma unroll
    for (int q = 0; q < 8; ++q) {
#pragma unroll
      for (int k0 = 0; k0 < 64; k0 += 32) {
        v16h a = ldsFragA(&h1[0][ks], HD1_, k0, lane);
        v16h b = ldsFragB(&slab[buf][q * 16][0], 64, k0, lane);
        acc[q] = WMMA_F16(a, b, acc[q]);
      }
    }
    __syncthreads();
  }
#pragma unroll
  for (int q = 0; q < 8; ++q) {
    int n = q * 16 + (lane & 15);
    int mB = (lane & 16) ? 8 : 0;
    float bv = sbd2[n];
#pragma unroll
    for (int r = 0; r < 8; ++r) {
      float v = acc[q][r] + bv;
      h2[mB + r][n] = (_Float16)(v > 0.f ? v : 0.f);
    }
  }
  __syncthreads();
  if (lane < 16) {
    int r = tile * 16 + lane;
    int n = r / T_, t = r - n * T_;
    float s = bd3[0];
    for (int j = 0; j < HD2_; ++j) s += (float)h2[lane][j] * swd3[j];
    float xr = s > 0.f ? s : 0.f;
    xrec[r] = xr;
    float mk = mask[(size_t)n * T_ + t];
    float dx = X[(size_t)r] - xr;
    red[lane] = dx * dx * mk;
  }
  __syncthreads();
  if (lane == 0) {
    float s = 0.f;
    for (int i = 0; i < 16; ++i) s += red[i];
    atomicAdd(&accum[0], s);
  }
}

// =================== KL term ===================================================
__global__ void kl_kernel(const float* mask, const float* zt, const float* logp, float* accum) {
  __shared__ float red[256];
  int idx = blockIdx.x * 256 + threadIdx.x;
  float contrib = 0.f;
  if (idx < N_ * T_) {
    float s = 0.f;
    for (int j = 0; j < 16; ++j) { float z = zt[(size_t)idx * 16 + j]; s += z * z; }
    float log_pzt = -0.5f * s - (float)QO_ * LOG2PI_;
    contrib = (logp[idx] - log_pzt) * mask[idx];
  }
  red[threadIdx.x] = contrib;
  __syncthreads();
  for (int off = 128; off > 0; off >>= 1) {
    if (threadIdx.x < off) red[threadIdx.x] += red[threadIdx.x + off];
    __syncthreads();
  }
  if (threadIdx.x == 0) atomicAdd(&accum[1], red[0]);
}

__global__ void fin_kernel(const float* accum, float* out) {
  if (threadIdx.x == 0) {
    out[N_ * T_]     = (float)NDATA_ * accum[0] / (float)N_;
    out[N_ * T_ + 1] = (float)NDATA_ * accum[1] / (float)N_;
  }
}

// =================== host launcher =============================================
extern "C" void kernel_launch(void* const* d_in, const int* in_sizes, int n_in,
                              void* d_out, int out_size, void* d_ws, size_t ws_size,
                              hipStream_t stream) {
  (void)in_sizes; (void)n_in; (void)out_size; (void)ws_size;
  const float* X    = (const float*)d_in[0];
  const float* cats = (const float*)d_in[1];
  const float* mask = (const float*)d_in[2];
  const float* eps  = (const float*)d_in[3];
  const float* We1  = (const float*)d_in[4];
  const float* be1  = (const float*)d_in[5];
  const float* Wm   = (const float*)d_in[6];
  const float* bm   = (const float*)d_in[7];
  const float* Wv   = (const float*)d_in[8];
  const float* bv   = (const float*)d_in[9];
  const float* Wode = (const float*)d_in[10];
  const float* bode = (const float*)d_in[11];
  const float* w1   = (const float*)d_in[12];
  const float* b1   = (const float*)d_in[13];
  const float* w2   = (const float*)d_in[14];
  const float* b2   = (const float*)d_in[15];
  const float* w3   = (const float*)d_in[16];
  const float* b3   = (const float*)d_in[17];
  const float* w4   = (const float*)d_in[18];
  const float* b4   = (const float*)d_in[19];
  const float* Wd1  = (const float*)d_in[20];
  const float* bd1  = (const float*)d_in[21];
  const float* Wd2  = (const float*)d_in[22];
  const float* bd2  = (const float*)d_in[23];
  const float* Wd3  = (const float*)d_in[24];
  const float* bd3  = (const float*)d_in[25];

  float* out = (float*)d_out;

  float*     zt    = (float*)d_ws;                       // N*T*16
  float*     logp  = zt + (size_t)N_ * T_ * 16;          // N*T
  float*     accum = logp + (size_t)N_ * T_;             // 16 (2 used)
  _Float16*  wd2t  = (_Float16*)(accum + 16);            // HD2*HD1 halves

  const size_t odeSmem = 104 * 1024;                     // carved usage ~99.8 KB

  zero_acc_kernel<<<1, 32, 0, stream>>>(accum);
  prep_wd2_kernel<<<(HD2_ * HD1_ + 255) / 256, 256, 0, stream>>>(Wd2, wd2t);
  enc_kernel<<<N_ / 16, 32, 0, stream>>>(X, cats, eps, We1, be1, Wm, bm, Wv, bv,
                                         Wode, bode, zt, logp);
  ode_kernel<<<N_ / 16, 32 * ODE_WAVES, odeSmem, stream>>>(w1, b1, w2, b2, w3, b3, w4, b4,
                                                           zt, logp);
  dec_kernel<<<(N_ * T_) / 16, 32, 0, stream>>>(X, cats, mask, Wd1, bd1, bd2, Wd3, bd3,
                                                wd2t, zt, out, accum);
  kl_kernel<<<(N_ * T_ + 255) / 256, 256, 0, stream>>>(mask, zt, logp, accum);
  fin_kernel<<<1, 32, 0, stream>>>(accum, out);
}
